// AdaMHF_49890340111125
// MI455X (gfx1250) — compile-verified, matrix operand order checked by reading, not run on hardware
//
#include <hip/hip_runtime.h>
#include <hip/hip_bf16.h>
#include <math.h>

// ---------------------------------------------------------------------------
// AdaMHF fused pipeline for MI455X (gfx1250, wave32, WMMA).
// Memory-bound workload: ~1GB token streaming dominates; GEMM chain is
// L2-resident. bf16 WMMA (v_wmma_f32_16x16x32_bf16) with f32 accumulate.
// ---------------------------------------------------------------------------

typedef __attribute__((ext_vector_type(16))) __bf16 v16bf;
typedef __attribute__((ext_vector_type(8)))  __bf16 v8bf;
typedef __attribute__((ext_vector_type(8)))  float  v8f;

static constexpr int kD  = 1024;
static constexpr int kDH = 4096;
static constexpr int kE  = 4;
static constexpr int kR  = 16;
static constexpr int kO  = 256;
static constexpr int kC  = 4;
static constexpr int kB  = 256;
static constexpr int kNP = 512;
static constexpr int kNG = 256;

enum { MODE_BIAS = 0, MODE_RELU = 1, MODE_ELU = 2, MODE_GATED = 3 };

// ---------------------------------------------------------------------------
// Kernel 1: token reductions.  top1 = max over tokens, avg = mean over
// tokens[1:].  Coalesced float4 streaming; grid (B, D/256), 64 threads.
// Writes bf16 top1 (GEMM-A input) and bf16 avg directly into the concat
// buffer's second half (cols D..2D-1).
// ---------------------------------------------------------------------------
__global__ __launch_bounds__(64) void token_reduce_kernel(
    const float* __restrict__ x, int Ntok,
    __bf16* __restrict__ top1,
    __bf16* __restrict__ avgOut, int avgLd)
{
    const int b = blockIdx.x;
    const int f = blockIdx.y * 256 + threadIdx.x * 4;
    const float* base = x + ((size_t)b * Ntok) * kD + f;

    float4 m = *(const float4*)base;
    float sx = 0.f, sy = 0.f, sz = 0.f, sw = 0.f;
    for (int n = 1; n < Ntok; ++n) {
        float4 v = *(const float4*)(base + (size_t)n * kD);
        m.x = fmaxf(m.x, v.x); m.y = fmaxf(m.y, v.y);
        m.z = fmaxf(m.z, v.z); m.w = fmaxf(m.w, v.w);
        sx += v.x; sy += v.y; sz += v.z; sw += v.w;
    }
    const float inv = 1.f / (float)(Ntok - 1);
    const size_t t0 = (size_t)b * kD + f;
    top1[t0 + 0] = (__bf16)m.x; top1[t0 + 1] = (__bf16)m.y;
    top1[t0 + 2] = (__bf16)m.z; top1[t0 + 3] = (__bf16)m.w;
    const size_t a0 = (size_t)b * avgLd + f;
    avgOut[a0 + 0] = (__bf16)(sx * inv); avgOut[a0 + 1] = (__bf16)(sy * inv);
    avgOut[a0 + 2] = (__bf16)(sz * inv); avgOut[a0 + 3] = (__bf16)(sw * inv);
}

// ---------------------------------------------------------------------------
// Kernel 2: fp32 -> bf16 weight convert + transpose.  in [K,N] row-major fp32
// -> out [N,K] row-major bf16 so WMMA B-fragments load with the same
// coalesced b128 pattern as A-fragments.  LDS 32x33 tile transpose.
// ---------------------------------------------------------------------------
__global__ __launch_bounds__(256) void convt_kernel(
    const float* __restrict__ in, long inZ, int rowStart, int K, int N,
    __bf16* __restrict__ out, long outZ)
{
    __shared__ float t[32][33];
    const int z = blockIdx.z;
    const float* src = in + (size_t)z * inZ + (size_t)rowStart * N;
    __bf16* dst = out + (size_t)z * outZ;
    const int k0 = blockIdx.y * 32, n0 = blockIdx.x * 32;
    const int x = threadIdx.x;
    for (int yy = threadIdx.y; yy < 32; yy += 8)
        t[yy][x] = src[(size_t)(k0 + yy) * N + n0 + x];
    __syncthreads();
    for (int yy = threadIdx.y; yy < 32; yy += 8)
        dst[(size_t)(n0 + yy) * K + k0 + x] = (__bf16)t[x][yy];
}

// ---------------------------------------------------------------------------
// Kernel 3: generic bf16 WMMA GEMM with fused epilogues.
//   C[M,N] = epilogue( A[M,K](bf16) @ Bt[N,K]^T(bf16) + bias[N] )
// Block = 128 threads = 4 waves; wave tile 32x32 (2x2 WMMA frags);
// block tile 64x64; grid (N/64, M/64, Z).
// Fragment layout per CDNA5 ISA (16-bit A 16x32): lane L(<16) holds row L,
// K 0..7 / 16..23; lane L+16 holds row L, K 8..15 / 24..31 -> two b128 loads.
// ---------------------------------------------------------------------------
__device__ __forceinline__ void gemm_epilogue(
    const v8f& c, int row0, int col,
    const float* __restrict__ bias,
    float* __restrict__ outF, __bf16* __restrict__ outB,
    const float* __restrict__ prev, const float* __restrict__ gate,
    int N, int mode)
{
    const float bi = bias ? bias[col] : 0.f;
#pragma unroll
    for (int i = 0; i < 8; ++i) {
        const int row = row0 + i;
        const size_t idx = (size_t)row * N + col;
        float v = c[i] + bi;
        if (mode == MODE_RELU) {
            v = fmaxf(v, 0.f);
        } else if (mode == MODE_ELU) {
            v = (v > 0.f) ? v : expm1f(v);
        } else if (mode == MODE_GATED) {
            const float e = (v > 0.f) ? v : expm1f(v);
            const float g = 1.f / (1.f + __expf(-gate[idx]));
            v = prev[idx] + g * e;
        }
        if (outF) outF[idx] = v;
        if (outB) outB[idx] = (__bf16)v;
    }
}

__global__ __launch_bounds__(128) void wmma_gemm_bf16(
    const __bf16* __restrict__ A, long aZ,
    const __bf16* __restrict__ Bt, long bZ,
    const float* __restrict__ bias, long biasZ,
    float* __restrict__ outF, long ofZ,
    __bf16* __restrict__ outB, long obZ,
    const float* __restrict__ prev, const float* __restrict__ gate,
    int K, int N, int mode)
{
    const int z = blockIdx.z;
    A  += (size_t)z * aZ;
    Bt += (size_t)z * bZ;
    const float* biasZp = bias ? bias + (size_t)z * biasZ : nullptr;
    float*  outFz = outF ? outF + (size_t)z * ofZ : nullptr;
    __bf16* outBz = outB ? outB + (size_t)z * obZ : nullptr;

    const int lane = threadIdx.x & 31;
    const int wave = threadIdx.x >> 5;
    const int m0 = blockIdx.y * 64 + (wave >> 1) * 32;
    const int n0 = blockIdx.x * 64 + (wave & 1) * 32;
    const int r  = lane & 15;
    const int hi = lane >> 4;

    const __bf16* a0p = A  + (size_t)(m0 + r)      * K + hi * 8;
    const __bf16* a1p = A  + (size_t)(m0 + 16 + r) * K + hi * 8;
    const __bf16* b0p = Bt + (size_t)(n0 + r)      * K + hi * 8;
    const __bf16* b1p = Bt + (size_t)(n0 + 16 + r) * K + hi * 8;

    v8f c00 = {0.f,0.f,0.f,0.f,0.f,0.f,0.f,0.f};
    v8f c01 = {0.f,0.f,0.f,0.f,0.f,0.f,0.f,0.f};
    v8f c10 = {0.f,0.f,0.f,0.f,0.f,0.f,0.f,0.f};
    v8f c11 = {0.f,0.f,0.f,0.f,0.f,0.f,0.f,0.f};

    for (int k = 0; k < K; k += 32) {
        union { v16bf v; v8bf h[2]; } a0, a1, b0, b1;
        a0.h[0] = *(const v8bf*)(a0p + k); a0.h[1] = *(const v8bf*)(a0p + k + 16);
        a1.h[0] = *(const v8bf*)(a1p + k); a1.h[1] = *(const v8bf*)(a1p + k + 16);
        b0.h[0] = *(const v8bf*)(b0p + k); b0.h[1] = *(const v8bf*)(b0p + k + 16);
        b1.h[0] = *(const v8bf*)(b1p + k); b1.h[1] = *(const v8bf*)(b1p + k + 16);
        // prefetch next weight chunk (streamed operand) -> global_prefetch_b8
        __builtin_prefetch(b0p + k + 128, 0, 3);
        __builtin_prefetch(b1p + k + 128, 0, 3);
        c00 = __builtin_amdgcn_wmma_f32_16x16x32_bf16(false, a0.v, false, b0.v, (short)0, c00, false, false);
        c01 = __builtin_amdgcn_wmma_f32_16x16x32_bf16(false, a0.v, false, b1.v, (short)0, c01, false, false);
        c10 = __builtin_amdgcn_wmma_f32_16x16x32_bf16(false, a1.v, false, b0.v, (short)0, c10, false, false);
        c11 = __builtin_amdgcn_wmma_f32_16x16x32_bf16(false, a1.v, false, b1.v, (short)0, c11, false, false);
    }

    // C/D layout: lane<16 -> col=lane, rows i; lane>=16 -> col=lane-16, rows i+8.
    gemm_epilogue(c00, m0 +      hi * 8, n0 +      r, biasZp, outFz, outBz, prev, gate, N, mode);
    gemm_epilogue(c01, m0 +      hi * 8, n0 + 16 + r, biasZp, outFz, outBz, prev, gate, N, mode);
    gemm_epilogue(c10, m0 + 16 + hi * 8, n0 +      r, biasZp, outFz, outBz, prev, gate, N, mode);
    gemm_epilogue(c11, m0 + 16 + hi * 8, n0 + 16 + r, biasZp, outFz, outBz, prev, gate, N, mode);
}

// ---------------------------------------------------------------------------
// Kernel 4: row softmax of m1 (f32 [B,D]) -> bf16 into concat[:, 0:D].
// ---------------------------------------------------------------------------
__global__ __launch_bounds__(256) void softmax_bf16_kernel(
    const float* __restrict__ m1, __bf16* __restrict__ out, int outLd)
{
    const int b = blockIdx.x, t = threadIdx.x;
    const float* row = m1 + (size_t)b * kD;
    float v0 = row[t * 4 + 0], v1 = row[t * 4 + 1];
    float v2 = row[t * 4 + 2], v3 = row[t * 4 + 3];
    __shared__ float red[256];
    red[t] = fmaxf(fmaxf(v0, v1), fmaxf(v2, v3));
    __syncthreads();
    for (int s = 128; s > 0; s >>= 1) {
        if (t < s) red[t] = fmaxf(red[t], red[t + s]);
        __syncthreads();
    }
    const float mx = red[0];
    __syncthreads();
    v0 = __expf(v0 - mx); v1 = __expf(v1 - mx);
    v2 = __expf(v2 - mx); v3 = __expf(v3 - mx);
    red[t] = v0 + v1 + v2 + v3;
    __syncthreads();
    for (int s = 128; s > 0; s >>= 1) {
        if (t < s) red[t] += red[t + s];
        __syncthreads();
    }
    const float inv = 1.f / red[0];
    const size_t o = (size_t)b * outLd + t * 4;
    out[o + 0] = (__bf16)(v0 * inv); out[o + 1] = (__bf16)(v1 * inv);
    out[o + 2] = (__bf16)(v2 * inv); out[o + 3] = (__bf16)(v3 * inv);
}

// ---------------------------------------------------------------------------
// Kernel 5: LMF rank combine + classifier.
// out[b,c] = sum_o ( sum_r fw[r]*Fa[r,b,o]*Fv[r,b,o] + fb[o] ) * W[o,c] + bcls[c]
// ---------------------------------------------------------------------------
__global__ __launch_bounds__(256) void lmf_combine_cls_kernel(
    const float* __restrict__ Fa, const float* __restrict__ Fv,
    const float* __restrict__ fw, const float* __restrict__ fb,
    const float* __restrict__ clsW, const float* __restrict__ clsB,
    float* __restrict__ out)
{
    const int b = blockIdx.x, o = threadIdx.x;
    float fused = fb[o];
#pragma unroll
    for (int r = 0; r < kR; ++r) {
        const size_t idx = ((size_t)r * kB + b) * kO + o;
        fused += fw[r] * Fa[idx] * Fv[idx];
    }
    __shared__ float red[kC][256];
#pragma unroll
    for (int c = 0; c < kC; ++c) red[c][o] = fused * clsW[o * kC + c];
    __syncthreads();
    for (int s = 128; s > 0; s >>= 1) {
        if (o < s) {
            red[0][o] += red[0][o + s];
            red[1][o] += red[1][o + s];
            red[2][o] += red[2][o + s];
            red[3][o] += red[3][o + s];
        }
        __syncthreads();
    }
    if (o < kC) out[b * kC + o] = red[o][0] + clsB[o];
}

// ---------------------------------------------------------------------------
// Host-side orchestration.
// ---------------------------------------------------------------------------
extern "C" void kernel_launch(void* const* d_in, const int* in_sizes, int n_in,
                              void* d_out, int out_size, void* d_ws, size_t ws_size,
                              hipStream_t stream)
{
    (void)in_sizes; (void)n_in; (void)out_size; (void)ws_size;

    // --- inputs (leaf order of setup_inputs dicts) ---
    const float* x_path = (const float*)d_in[0];
    const float* x_omic = (const float*)d_in[1];
    const float* ap_w1 = (const float*)d_in[2];  const float* ap_b1 = (const float*)d_in[3];
    const float* ap_wc = (const float*)d_in[4];  const float* ap_bc = (const float*)d_in[5];
    const float* ap_w2 = (const float*)d_in[6];  const float* ap_b2 = (const float*)d_in[7];
    const float* ag_w1 = (const float*)d_in[8];  const float* ag_b1 = (const float*)d_in[9];
    const float* ag_wc = (const float*)d_in[10]; const float* ag_bc = (const float*)d_in[11];
    const float* ag_w2 = (const float*)d_in[12]; const float* ag_b2 = (const float*)d_in[13];
    const float* pp_wf1 = (const float*)d_in[14]; const float* pp_bf1 = (const float*)d_in[15];
    const float* pp_wf2 = (const float*)d_in[16]; const float* pp_bf2 = (const float*)d_in[17];
    const float* pp_ew1 = (const float*)d_in[18]; const float* pp_eb1 = (const float*)d_in[19];
    const float* pp_ew2 = (const float*)d_in[20]; const float* pp_eb2 = (const float*)d_in[21];
    const float* pg_wf1 = (const float*)d_in[22]; const float* pg_bf1 = (const float*)d_in[23];
    const float* pg_wf2 = (const float*)d_in[24]; const float* pg_bf2 = (const float*)d_in[25];
    const float* pg_ew1 = (const float*)d_in[26]; const float* pg_eb1 = (const float*)d_in[27];
    const float* pg_ew2 = (const float*)d_in[28]; const float* pg_eb2 = (const float*)d_in[29];
    const float* lmf_fa = (const float*)d_in[30]; const float* lmf_fv = (const float*)d_in[31];
    const float* lmf_fw = (const float*)d_in[32]; const float* lmf_fb = (const float*)d_in[33];
    const float* cls_w  = (const float*)d_in[34]; const float* cls_b  = (const float*)d_in[35];

    // --- workspace layout (256B aligned bump allocator) ---
    char* wsp = (char*)d_ws;
    size_t off = 0;
    auto alloc = [&](size_t bytes) -> void* {
        void* p = wsp + off;
        off = (off + bytes + 255) & ~(size_t)255;
        return p;
    };
    // converted bf16 weights, transposed [N,K]
    __bf16* w1t_p  = (__bf16*)alloc(sizeof(__bf16) * kD * kD);
    __bf16* wct_p  = (__bf16*)alloc(sizeof(__bf16) * kD * 2 * kD);
    __bf16* w2t_p  = (__bf16*)alloc(sizeof(__bf16) * kD * kD);
    __bf16* w1t_g  = (__bf16*)alloc(sizeof(__bf16) * kD * kD);
    __bf16* wct_g  = (__bf16*)alloc(sizeof(__bf16) * kD * 2 * kD);
    __bf16* w2t_g  = (__bf16*)alloc(sizeof(__bf16) * kD * kD);
    __bf16* wf1t_p = (__bf16*)alloc(sizeof(__bf16) * (size_t)kDH * kD);
    __bf16* wf2t_p = (__bf16*)alloc(sizeof(__bf16) * (size_t)kD * kDH);
    __bf16* ew1t_p = (__bf16*)alloc(sizeof(__bf16) * (size_t)kE * kD * kD);
    __bf16* ew2t_p = (__bf16*)alloc(sizeof(__bf16) * (size_t)kE * kD * kD);
    __bf16* wf1t_g = (__bf16*)alloc(sizeof(__bf16) * (size_t)kDH * kD);
    __bf16* wf2t_g = (__bf16*)alloc(sizeof(__bf16) * (size_t)kD * kDH);
    __bf16* ew1t_g = (__bf16*)alloc(sizeof(__bf16) * (size_t)kE * kD * kD);
    __bf16* ew2t_g = (__bf16*)alloc(sizeof(__bf16) * (size_t)kE * kD * kD);
    __bf16* fat    = (__bf16*)alloc(sizeof(__bf16) * (size_t)kR * kO * kD);
    __bf16* fvt    = (__bf16*)alloc(sizeof(__bf16) * (size_t)kR * kO * kD);
    // activations
    __bf16* top1_p  = (__bf16*)alloc(sizeof(__bf16) * kB * kD);
    __bf16* top1_g  = (__bf16*)alloc(sizeof(__bf16) * kB * kD);
    __bf16* concatP = (__bf16*)alloc(sizeof(__bf16) * kB * 2 * kD);
    __bf16* concatG = (__bf16*)alloc(sizeof(__bf16) * kB * 2 * kD);
    float*  m1_p    = (float*)alloc(sizeof(float) * kB * kD);
    float*  m1_g    = (float*)alloc(sizeof(float) * kB * kD);
    __bf16* c_p     = (__bf16*)alloc(sizeof(__bf16) * kB * kD);
    __bf16* c_g     = (__bf16*)alloc(sizeof(__bf16) * kB * kD);
    float*  h_pF    = (float*)alloc(sizeof(float) * kB * kD);
    __bf16* h_pB    = (__bf16*)alloc(sizeof(__bf16) * kB * kD);
    float*  h_gF    = (float*)alloc(sizeof(float) * kB * kD);
    __bf16* h_gB    = (__bf16*)alloc(sizeof(__bf16) * kB * kD);
    __bf16* fm_p    = (__bf16*)alloc(sizeof(__bf16) * (size_t)kB * kDH);
    __bf16* fm_g    = (__bf16*)alloc(sizeof(__bf16) * (size_t)kB * kDH);
    float*  out_pF  = (float*)alloc(sizeof(float) * kB * kD);
    __bf16* out_pB  = (__bf16*)alloc(sizeof(__bf16) * kB * kD);
    float*  out_gF  = (float*)alloc(sizeof(float) * kB * kD);
    __bf16* out_gB  = (__bf16*)alloc(sizeof(__bf16) * kB * kD);
    __bf16* hx_p    = (__bf16*)alloc(sizeof(__bf16) * kB * kD);
    __bf16* hx_g    = (__bf16*)alloc(sizeof(__bf16) * kB * kD);
    float*  Fa      = (float*)alloc(sizeof(float) * (size_t)kR * kB * kO);
    float*  Fv      = (float*)alloc(sizeof(float) * (size_t)kR * kB * kO);

    auto convt = [&](const float* in, long inZ, int rowStart, int K, int N,
                     __bf16* out, long outZ, int Z) {
        dim3 g(N / 32, K / 32, Z), b(32, 8);
        convt_kernel<<<g, b, 0, stream>>>(in, inZ, rowStart, K, N, out, outZ);
    };
    auto gemm = [&](const __bf16* A, long aZ, const __bf16* Bt, long bZ,
                    const float* bias, long biasZ,
                    float* outF, long ofZ, __bf16* outB, long obZ,
                    const float* prev, const float* gate,
                    int M, int N, int K, int mode, int Z) {
        dim3 g(N / 64, M / 64, Z), b(128);
        wmma_gemm_bf16<<<g, b, 0, stream>>>(A, aZ, Bt, bZ, bias, biasZ,
                                            outF, ofZ, outB, obZ, prev, gate,
                                            K, N, mode);
    };

    // --- weight conversion (one pass; everything after lives in L2) ---
    convt(ap_w1, 0, 0, kD, kD, w1t_p, 0, 1);
    convt(ap_wc, 0, 0, 2 * kD, kD, wct_p, 0, 1);
    convt(ap_w2, 0, 0, kD, kD, w2t_p, 0, 1);
    convt(ag_w1, 0, 0, kD, kD, w1t_g, 0, 1);
    convt(ag_wc, 0, 0, 2 * kD, kD, wct_g, 0, 1);
    convt(ag_w2, 0, 0, kD, kD, w2t_g, 0, 1);
    convt(pp_wf1, 0, 0, kD, kDH, wf1t_p, 0, 1);
    convt(pp_wf2, 0, 0, kDH, kD, wf2t_p, 0, 1);
    convt(pp_ew1, (long)kD * kD, 0, kD, kD, ew1t_p, (long)kD * kD, kE);
    convt(pp_ew2, (long)kD * kD, 0, kD, kD, ew2t_p, (long)kD * kD, kE);
    convt(pg_wf1, 0, 0, kD, kDH, wf1t_g, 0, 1);
    convt(pg_wf2, 0, 0, kDH, kD, wf2t_g, 0, 1);
    convt(pg_ew1, (long)kD * kD, 0, kD, kD, ew1t_g, (long)kD * kD, kE);
    convt(pg_ew2, (long)kD * kD, 0, kD, kD, ew2t_g, (long)kD * kD, kE);
    convt(lmf_fa, 1025L * kO, 1, kD, kO, fat, (long)kO * kD, kR);
    convt(lmf_fv, 1025L * kO, 1, kD, kO, fvt, (long)kO * kD, kR);

    // --- token reductions (bandwidth-critical) ---
    {
        dim3 g(kB, kD / 256), b(64);
        token_reduce_kernel<<<g, b, 0, stream>>>(x_path, kNP, top1_p, concatP + kD, 2 * kD);
        token_reduce_kernel<<<g, b, 0, stream>>>(x_omic, kNG, top1_g, concatG + kD, 2 * kD);
    }

    // --- ATSA (both modalities) ---
    gemm(top1_p, 0, w1t_p, 0, ap_b1, 0, m1_p, 0, nullptr, 0, nullptr, nullptr,
         kB, kD, kD, MODE_RELU, 1);
    gemm(top1_g, 0, w1t_g, 0, ag_b1, 0, m1_g, 0, nullptr, 0, nullptr, nullptr,
         kB, kD, kD, MODE_RELU, 1);
    softmax_bf16_kernel<<<kB, 256, 0, stream>>>(m1_p, concatP, 2 * kD);
    softmax_bf16_kernel<<<kB, 256, 0, stream>>>(m1_g, concatG, 2 * kD);
    gemm(concatP, 0, wct_p, 0, ap_bc, 0, nullptr, 0, c_p, 0, nullptr, nullptr,
         kB, kD, 2 * kD, MODE_BIAS, 1);
    gemm(concatG, 0, wct_g, 0, ag_bc, 0, nullptr, 0, c_g, 0, nullptr, nullptr,
         kB, kD, 2 * kD, MODE_BIAS, 1);
    gemm(c_p, 0, w2t_p, 0, ap_b2, 0, h_pF, 0, h_pB, 0, nullptr, nullptr,
         kB, kD, kD, MODE_RELU, 1);
    gemm(c_g, 0, w2t_g, 0, ag_b2, 0, h_gF, 0, h_gB, 0, nullptr, nullptr,
         kB, kD, kD, MODE_RELU, 1);

    // --- PREEG pathology (gate = sigmoid(h_g)) ---
    gemm(h_pB, 0, wf1t_p, 0, pp_bf1, 0, nullptr, 0, fm_p, 0, nullptr, nullptr,
         kB, kDH, kD, MODE_RELU, 1);
    gemm(fm_p, 0, wf2t_p, 0, pp_bf2, 0, out_pF, 0, out_pB, 0, nullptr, nullptr,
         kB, kD, kDH, MODE_BIAS, 1);
    for (int e = 0; e < kE; ++e) {
        gemm(out_pB, 0, ew1t_p + (size_t)e * kD * kD, 0, pp_eb1 + e * kD, 0,
             nullptr, 0, hx_p, 0, nullptr, nullptr, kB, kD, kD, MODE_ELU, 1);
        gemm(hx_p, 0, ew2t_p + (size_t)e * kD * kD, 0, pp_eb2 + e * kD, 0,
             out_pF, 0, out_pB, 0, out_pF, h_gF, kB, kD, kD, MODE_GATED, 1);
    }
    // --- PREEG genomic (gate = sigmoid(h_p)) ---
    gemm(h_gB, 0, wf1t_g, 0, pg_bf1, 0, nullptr, 0, fm_g, 0, nullptr, nullptr,
         kB, kDH, kD, MODE_RELU, 1);
    gemm(fm_g, 0, wf2t_g, 0, pg_bf2, 0, out_gF, 0, out_gB, 0, nullptr, nullptr,
         kB, kD, kDH, MODE_BIAS, 1);
    for (int e = 0; e < kE; ++e) {
        gemm(out_gB, 0, ew1t_g + (size_t)e * kD * kD, 0, pg_eb1 + e * kD, 0,
             nullptr, 0, hx_g, 0, nullptr, nullptr, kB, kD, kD, MODE_ELU, 1);
        gemm(hx_g, 0, ew2t_g + (size_t)e * kD * kD, 0, pg_eb2 + e * kD, 0,
             out_gF, 0, out_gB, 0, out_gF, h_pF, kB, kD, kD, MODE_GATED, 1);
    }

    // --- LMF: batched over rank (grid.z = R); ones-column folded into bias
    //     (bias[r] = fa[r, 0, :]).  _lmf(p, a=o_g, v=o_p).
    gemm(out_gB, 0, fat, (long)kO * kD, lmf_fa, 1025L * kO,
         Fa, (long)kB * kO, nullptr, 0, nullptr, nullptr,
         kB, kO, kD, MODE_BIAS, kR);
    gemm(out_pB, 0, fvt, (long)kO * kD, lmf_fv, 1025L * kO,
         Fv, (long)kB * kO, nullptr, 0, nullptr, nullptr,
         kB, kO, kD, MODE_BIAS, kR);

    // --- rank combine + classifier ---
    lmf_combine_cls_kernel<<<kB, 256, 0, stream>>>(Fa, Fv, lmf_fw, lmf_fb,
                                                   cls_w, cls_b, (float*)d_out);
}
